// SOM_61753039782082
// MI455X (gfx1250) — matrix operand order repaired and checked by reading, not compiled, not run on hardware
//
#include <hip/hip_runtime.h>

#define D_DIM     784
#define KP        800           // K padded to 25 * 32
#define KSTEPS    (KP / 32)     // 25
#define KCH       (KP / 8)      // 100 8-element chunks per row
#define NN        10000
#define NTILES    625           // 10000 / 16
#define TPB_TILES 25            // n-tiles per block (625 / 25 splits)
#define NSPLITS   25
#define MROWS     64            // batch rows per block
#define MTILES    64            // 4096 / 64
#define BATCH_N   4096

typedef __attribute__((ext_vector_type(16))) __bf16    v16bf;
typedef __attribute__((ext_vector_type(8)))  float     v8f;
typedef __attribute__((ext_vector_type(4)))  unsigned  u32x4;
typedef __attribute__((ext_vector_type(4)))  float     f32x4;

__device__ __forceinline__ unsigned short f2bf(float f) {
    unsigned u = __float_as_uint(f);
    u += 0x7FFFu + ((u >> 16) & 1u);            // round-to-nearest-even
    return (unsigned short)(u >> 16);
}
__device__ __forceinline__ float bf2f(unsigned short h) {
    return __uint_as_float(((unsigned)h) << 16);
}
// monotone float -> u32 key (smaller float => smaller key)
__device__ __forceinline__ unsigned fkey(float f) {
    unsigned u = __float_as_uint(f);
    return (u & 0x80000000u) ? ~u : (u | 0x80000000u);
}
// convert 8 fp32 -> 8 bf16 hi (packed u32x4) + 8 bf16 residual lo (packed u32x4)
__device__ __forceinline__ void cvt8(f32x4 a, f32x4 b, u32x4& hi, u32x4& lo) {
    unsigned short h[8], l[8];
    float f[8] = {a.x, a.y, a.z, a.w, b.x, b.y, b.z, b.w};
#pragma unroll
    for (int e = 0; e < 8; ++e) {
        h[e] = f2bf(f[e]);
        l[e] = f2bf(f[e] - bf2f(h[e]));
    }
    hi = (u32x4){(unsigned)h[0] | ((unsigned)h[1] << 16),
                 (unsigned)h[2] | ((unsigned)h[3] << 16),
                 (unsigned)h[4] | ((unsigned)h[5] << 16),
                 (unsigned)h[6] | ((unsigned)h[7] << 16)};
    lo = (u32x4){(unsigned)l[0] | ((unsigned)l[1] << 16),
                 (unsigned)l[2] | ((unsigned)l[3] << 16),
                 (unsigned)l[4] | ((unsigned)l[5] << 16),
                 (unsigned)l[6] | ((unsigned)l[7] << 16)};
}

// ---------------- prep: w_sq (fp32, exact) + init packed argmin slots --------
__global__ void som_bmu_prep(const float* __restrict__ W,
                             float* __restrict__ wsq,
                             unsigned long long* __restrict__ packed) {
    const int g = blockIdx.x * 256 + threadIdx.x;
    if (g < BATCH_N) packed[g] = ~0ull;
    const int lane = threadIdx.x & 31;
    const int row  = blockIdx.x * 8 + (threadIdx.x >> 5);
    if (row < NN) {
        float s = 0.f;
        for (int k = lane; k < D_DIM; k += 32) {
            float v = W[(size_t)row * D_DIM + k];
            s += v * v;
        }
#pragma unroll
        for (int m = 16; m >= 1; m >>= 1) s += __shfl_xor(s, m, 32);
        if (lane == 0) wsq[row] = s;
    }
}

// ---------------- main: split-bf16 WMMA GEMM + fused argmin ------------------
__global__ __launch_bounds__(256)
void som_bmu_main(const float* __restrict__ X, const float* __restrict__ W,
                  const float* __restrict__ wsq,
                  unsigned long long* __restrict__ packed) {
    __shared__ unsigned short Ah[MROWS * KP];   // 102400 B
    __shared__ unsigned short Al[MROWS * KP];   // 102400 B
    __shared__ unsigned short Bh[32 * KP];      //  51200 B
    __shared__ unsigned short Bl[32 * KP];      //  51200 B  -> 307200 total

    const int tid  = threadIdx.x;
    const int lane = tid & 31;
    const int wave = tid >> 5;
    const int wgrp = wave >> 2;    // which of the 2 staged n-tiles
    const int wrow = wave & 3;     // which 16-row M subtile

    u32x4* __restrict__ Ah4 = (u32x4*)Ah;
    u32x4* __restrict__ Al4 = (u32x4*)Al;
    u32x4* __restrict__ Bh4 = (u32x4*)Bh;
    u32x4* __restrict__ Bl4 = (u32x4*)Bl;

    // ---- stage A tile once: fp32 -> bf16 hi + residual lo, 8-wide chunks ----
    const int mbase = blockIdx.y * MROWS;
    for (int i = tid; i < MROWS * KCH; i += 256) {
        int r = i / KCH, ch = i - r * KCH;          // chunk ch covers cols ch*8..+7
        u32x4 hi = {}, lo = {};
        if (ch * 8 < D_DIM) {                       // 784 % 8 == 0: chunk fully valid
            const f32x4* xp = (const f32x4*)(X + (size_t)(mbase + r) * D_DIM + ch * 8);
            cvt8(xp[0], xp[1], hi, lo);
        }
        Ah4[i] = hi;                                 // uint4 index == chunk index
        Al4[i] = lo;
    }

    // per-lane fragment addresses (ISA 16-bit A 16x32 / B 32x16 layouts)
    const int aoff = (wrow * 16 + (lane & 15)) * KP + ((lane >> 4) * 8);
    const int boff = (wgrp * 16 + (lane & 15)) * KP + ((lane >> 4) * 8);

    unsigned long long mp[8];
#pragma unroll
    for (int r = 0; r < 8; ++r) mp[r] = ~0ull;

    const int tbegin = blockIdx.x * TPB_TILES;
    const int tend   = tbegin + TPB_TILES;          // <= NTILES by construction

    for (int j = 0; j < TPB_TILES; j += 2) {
        const int t0 = tbegin + j;
        __syncthreads();                            // prior compute done before overwrite
        // stage two 16-neuron W tiles, hi/lo bf16, 8-wide vectorized
        for (int i = tid; i < 32 * KCH; i += 256) {
            int r = i / KCH, ch = i - r * KCH;
            int t = t0 + (r >> 4);
            u32x4 hi = {}, lo = {};
            if (t < tend && ch * 8 < D_DIM) {
                const f32x4* wp =
                    (const f32x4*)(W + (size_t)(t * 16 + (r & 15)) * D_DIM + ch * 8);
                cvt8(wp[0], wp[1], hi, lo);
            }
            Bh4[i] = hi;
            Bl4[i] = lo;
        }
        __syncthreads();

        // prefetch next iteration's W tile-pair toward cache while we compute
        if (t0 + 3 < tend) {
            const char* nxt = (const char*)(W + (size_t)(t0 + 2) * 16 * D_DIM);
            for (int off = tid * 128; off < 32 * D_DIM * 4; off += 256 * 128)
                __builtin_prefetch(nxt + off, 0, 1);
        }

        const int myt = t0 + wgrp;                  // wave-uniform -> EXEC all ones
        if (myt < tend) {
            v8f accHH = {}, accHL = {}, accLH = {}; // 3 independent WMMA chains
            const u32x4* __restrict__ aph = (const u32x4*)Ah + (aoff >> 3);
            const u32x4* __restrict__ apl = (const u32x4*)Al + (aoff >> 3);
            const u32x4* __restrict__ bph = (const u32x4*)Bh + (boff >> 3);
            const u32x4* __restrict__ bpl = (const u32x4*)Bl + (boff >> 3);
            for (int kk = 0; kk < KSTEPS; ++kk) {
                union { u32x4 q[2]; v16bf v; } fah, fal, fbh, fbl;
                fah.q[0] = aph[kk * 4 + 0]; fah.q[1] = aph[kk * 4 + 2];
                fal.q[0] = apl[kk * 4 + 0]; fal.q[1] = apl[kk * 4 + 2];
                fbh.q[0] = bph[kk * 4 + 0]; fbh.q[1] = bph[kk * 4 + 2];
                fbl.q[0] = bpl[kk * 4 + 0]; fbl.q[1] = bpl[kk * 4 + 2];
                // cross ~= xh*wh + xh*wl + xl*wh   (fp32 accumulate)
                accHH = __builtin_amdgcn_wmma_f32_16x16x32_bf16(
                            false, fah.v, false, fbh.v, (short)0, accHH, false, false);
                accHL = __builtin_amdgcn_wmma_f32_16x16x32_bf16(
                            false, fah.v, false, fbl.v, (short)0, accHL, false, false);
                accLH = __builtin_amdgcn_wmma_f32_16x16x32_bf16(
                            false, fal.v, false, fbh.v, (short)0, accLH, false, false);
            }
            const int   n  = myt * 16 + (lane & 15);
            const float ws = wsq[n];
#pragma unroll
            for (int r = 0; r < 8; ++r) {
                float cross = accHH[r] + accHL[r] + accLH[r];
                float score = ws - 2.0f * cross;    // x_sq constant per row: dropped
                unsigned long long cand =
                    ((unsigned long long)fkey(score) << 32) | (unsigned)n;
                if (cand < mp[r]) mp[r] = cand;     // ties -> smallest n (jnp.argmin)
            }
        }
    }

    // reduce across the 16 lanes of each half (C layout: halves are disjoint rows)
#pragma unroll
    for (int r = 0; r < 8; ++r) {
        unsigned long long v = mp[r];
#pragma unroll
        for (int m = 1; m <= 8; m <<= 1) {
            unsigned lo = __shfl_xor((unsigned)v, m, 32);
            unsigned hi = __shfl_xor((unsigned)(v >> 32), m, 32);
            unsigned long long o = ((unsigned long long)hi << 32) | lo;
            if (o < v) v = o;
        }
        mp[r] = v;
    }
    if ((lane & 15) == 0) {                         // lanes 0 (rows r) and 16 (rows 8+r)
        const int rowbase = mbase + wrow * 16 + ((lane >> 4) * 8);
#pragma unroll
        for (int r = 0; r < 8; ++r)
            atomicMin(&packed[rowbase + r], mp[r]);
    }
}

// ---------------- gather: out[b,:] = W[argmin,:] -----------------------------
__global__ void som_bmu_gather(const float* __restrict__ W,
                               const unsigned long long* __restrict__ packed,
                               float* __restrict__ out) {
    const int row = blockIdx.x;
    unsigned idx = (unsigned)packed[row];           // low 32 bits = neuron index
    if (idx >= (unsigned)NN) idx = 0;               // safety clamp
    const f32x4* __restrict__ src = (const f32x4*)(W + (size_t)idx * D_DIM);
    f32x4* __restrict__ dst = (f32x4*)(out + (size_t)row * D_DIM);
    for (int i = threadIdx.x; i < D_DIM / 4; i += 256) dst[i] = src[i];
}

extern "C" void kernel_launch(void* const* d_in, const int* in_sizes, int n_in,
                              void* d_out, int out_size, void* d_ws, size_t ws_size,
                              hipStream_t stream) {
    const float* X = (const float*)d_in[0];         // (4096, 784) fp32
    const float* W = (const float*)d_in[1];         // (10000, 784) fp32
    float* out = (float*)d_out;                     // (4096, 784) fp32

    unsigned long long* packed = (unsigned long long*)d_ws;                    // 4096 u64
    float* wsq = (float*)((char*)d_ws + BATCH_N * sizeof(unsigned long long)); // 10000 f32

    som_bmu_prep<<<(NN + 7) / 8, 256, 0, stream>>>(W, wsq, packed);

    dim3 grid(NSPLITS, MTILES);                     // 25 x 64 = 1600 blocks, 8 waves each
    som_bmu_main<<<grid, 256, 0, stream>>>(X, W, wsq, packed);

    som_bmu_gather<<<BATCH_N, 256, 0, stream>>>(W, packed, out);
}
